// WindowAttention_76716705841252
// MI455X (gfx1250) — matrix operand order repaired and checked by reading, not compiled, not run on hardware
//
#include <hip/hip_runtime.h>
#include <hip/hip_bf16.h>

typedef __attribute__((ext_vector_type(16))) _Float16 v16h;
typedef __attribute__((ext_vector_type(8)))  float    v8f;

#define XT_LD 132   // Xt row stride (halfs), 64 x 128 payload
#define QK_LD 260   // QKt row stride, 64 x 256 payload (Q^T | K^T)
#define V_LD   68   // V row stride, 128 x 64 payload
#define S_LD   68   // per-head score row stride, 64 x 64 payload
#define OT_LD 132   // Ot row stride, 64 x 128 payload
#define Y_LD   68   // Yf (fp32 result) row stride, 128 x 64 payload

// byte offsets inside the single shared block
#define OFF_XT   0
#define OFF_QKT  16896    // 64*132*2
#define OFF_VB   50176    // + 64*260*2
#define OFF_SH   67584    // + 128*68*2
#define OFF_OT   102400   // + 4*64*68*2
#define OFF_BSH  119296   // + 64*132*2
#define SMEM_SZ  119808   // + 128*4

union FragU { v16h v; unsigned int u[8]; };

// A-matrix 16x32 f16 fragment (ISA 7.12.2): lane m=lane&15, half=lane>>4,
// VGPR j in 0..3 -> k = half*8 + 2j ; j in 4..7 -> k = 16 + half*8 + 2(j-4)
__device__ __forceinline__ v16h load_frag_a(const _Float16* p, int row0, int col0,
                                            int ld, int lane) {
  const int m = lane & 15, half = lane >> 4;
  const _Float16* base = p + (size_t)(row0 + m) * ld + col0;
  FragU f;
#pragma unroll
  for (int j = 0; j < 8; ++j) {
    const int k = ((j >> 2) << 4) + half * 8 + ((j & 3) << 1);
    f.u[j] = *(const unsigned int*)(base + k);
  }
  return f.v;
}

// B-matrix 32x16 f16 fragment, sourced from B^T stored row-major [N][K]:
// lane n=lane&15, half=lane>>4, VGPR j in 0..3 -> k = half*8 + 2j ; j in 4..7 -> k = 16 + half*8 + 2(j-4)
__device__ __forceinline__ v16h load_frag_b(const _Float16* p, int n0, int k0,
                                            int ld, int lane) {
  const int n = lane & 15, half = lane >> 4;
  const _Float16* base = p + (size_t)(n0 + n) * ld + k0 + half * 8;
  FragU f;
#pragma unroll
  for (int j = 0; j < 8; ++j) f.u[j] = *(const unsigned int*)(base + ((j >> 2) << 4) + ((j & 3) << 1));
  return f.v;
}

__device__ __forceinline__ v8f wmma_f16(v16h a, v16h b, v8f c) {
  return __builtin_amdgcn_wmma_f32_16x16x32_f16(false, a, false, b,
                                                (short)0, c, false, false);
}

// D-tile per-lane row store: 8 consecutive f16 (8B aligned) as 2 x b64
__device__ __forceinline__ void store_row8_f16(_Float16* dst, const v8f acc,
                                               float mul) {
  union { uint2 u2[2]; _Float16 h[8]; } pk;
#pragma unroll
  for (int r = 0; r < 8; ++r) pk.h[r] = (_Float16)(acc[r] * mul);
  *(uint2*)(dst)     = pk.u2[0];
  *(uint2*)(dst + 4) = pk.u2[1];
}

// D-tile per-lane row store: 8 consecutive f32 (16B aligned) as 2 x b128
__device__ __forceinline__ void store_row8_f32(float* dst, const v8f acc,
                                               float badd) {
  float4 a; a.x = acc[0] + badd; a.y = acc[1] + badd; a.z = acc[2] + badd; a.w = acc[3] + badd;
  float4 b; b.x = acc[4] + badd; b.y = acc[5] + badd; b.z = acc[6] + badd; b.w = acc[7] + badd;
  *(float4*)(dst)     = a;
  *(float4*)(dst + 4) = b;
}

// One-shot fp32 -> f16 conversion of both weight matrices into workspace.
__global__ void cvt_weights_kernel(const float* __restrict__ wq,
                                   const float* __restrict__ wp,
                                   _Float16* __restrict__ out) {
  const int i = blockIdx.x * blockDim.x + threadIdx.x;
  if (i < 384 * 128) out[i] = (_Float16)wq[i];
  if (i < 128 * 128) out[384 * 128 + i] = (_Float16)wp[i];
}

__global__ __launch_bounds__(256)
void win_attn_kernel(const float* __restrict__ x,
                     const _Float16* __restrict__ wq16,   // [384][128] f16
                     const _Float16* __restrict__ wp16,   // [128][128] f16
                     const float* __restrict__ bias,      // [128] f32
                     float* __restrict__ out) {
  __shared__ __align__(16) unsigned char smem[SMEM_SZ];
  _Float16* Xt  = (_Float16*)(smem + OFF_XT);   // [n][c]
  _Float16* QKt = (_Float16*)(smem + OFF_QKT);  // [n][0..127]=Q^T(scaled), [128..255]=K^T
  _Float16* Vb  = (_Float16*)(smem + OFF_VB);   // [c][n]
  _Float16* Sh  = (_Float16*)(smem + OFF_SH);   // attn: [h][q][key], key contiguous
  _Float16* Ot  = (_Float16*)(smem + OFF_OT);   // [n][c]
  float*    bsh = (float*)(smem + OFF_BSH);
  float*    Yf  = (float*)(smem + OFF_XT);      // aliases Xt/QKt (dead by stage 5)

  const int tid  = threadIdx.x;
  const int lane = tid & 31;
  const int half = lane >> 4;
  // force wave index into an SGPR: tile loops & stage branches become scalar
  const int wave = __builtin_amdgcn_readfirstlane(tid >> 5);

  const int wd  = blockIdx.x;          // 8192 windows
  const int b   = wd >> 10;            // 1024 windows per batch image
  const int rem = wd & 1023;
  const int h0  = (rem >> 5) << 3;
  const int w0  = (rem & 31) << 3;
  const float* xb = x + (size_t)b * 128 * 65536;

  if (tid < 128) bsh[tid] = bias[tid];

  // ---- Stage 0: load x window [128][8x8] as float4, store f16 Xt[n][c] ----
#pragma unroll
  for (int it = 0; it < 8; ++it) {
    const int q  = it * 256 + tid;      // 2048 float4 quads
    const int c  = q >> 4;              // 0..127
    const int n0 = (q & 15) << 2;       // token base, multiple of 4
    const int i = n0 >> 3, j0 = n0 & 7;
    const float4 v = *(const float4*)(xb + (size_t)c * 65536 +
                                      (size_t)(h0 + i) * 256 + (w0 + j0));
    Xt[(n0 + 0) * XT_LD + c] = (_Float16)v.x;
    Xt[(n0 + 1) * XT_LD + c] = (_Float16)v.y;
    Xt[(n0 + 2) * XT_LD + c] = (_Float16)v.z;
    Xt[(n0 + 3) * XT_LD + c] = (_Float16)v.w;
  }
  __syncthreads();

  const float scale = 0.17677669529663687f;  // 1/sqrt(32)

  // ---- Stage 1a: Q,K = w_qk[256x128] @ xw[128x64]; M=channel, 16x4 tiles ----
  // D rows (M=ch) are contiguous in QKt[token][ch] -> 2 x b64 store per lane.
  for (int t = wave; t < 64; t += 8) {
    const int mt = t >> 2, nt = t & 3;           // mt: ch-tile 0..15, nt: token-tile
    v8f acc = {};
#pragma unroll
    for (int ks = 0; ks < 4; ++ks) {
      v16h a  = load_frag_a(wq16, 16 * mt, 32 * ks, 128, lane);
      v16h bf = load_frag_b(Xt, 16 * nt, 32 * ks, XT_LD, lane);
      acc = wmma_f16(a, bf, acc);
    }
    const int tok = 16 * nt + (lane & 15);
    _Float16* dst = QKt + tok * QK_LD + 16 * mt + 8 * half;
    store_row8_f16(dst, acc, (mt < 8) ? scale : 1.0f);   // fold softmax scale into Q
  }

  // ---- Stage 1b: V^T = xw^T[64x128] @ w_v^T[128x128]; M=token, 4x8 tiles ----
  // B^T = w_qkv rows 256..383 as stored ([o][c]). D rows (M=token) contiguous
  // in Vb[ch][token].
  for (int t = wave; t < 32; t += 8) {
    const int nt = t >> 2, mt = t & 3;           // nt: vch-tile 0..7, mt: token-tile
    v8f acc = {};
#pragma unroll
    for (int ks = 0; ks < 4; ++ks) {
      v16h a  = load_frag_a(Xt, 16 * mt, 32 * ks, XT_LD, lane);
      v16h bf = load_frag_b(wq16, 256 + 16 * nt, 32 * ks, 128, lane);
      acc = wmma_f16(a, bf, acc);
    }
    const int vch = 16 * nt + (lane & 15);
    _Float16* dst = Vb + vch * V_LD + 16 * mt + 8 * half;
    store_row8_f16(dst, acc, 1.0f);
  }
  __syncthreads();

  // ---- Stage 2: S^T = K^T Q per head; M=key, N=query; 4 heads x 4x4 tiles ----
  // D rows (M=key) contiguous in Sh[q][key].
  for (int t = wave; t < 64; t += 8) {
    const int h  = t >> 4;
    const int mt = (t >> 2) & 3, nt = t & 3;     // mt: key-tile, nt: query-tile
    v16h a  = load_frag_a(QKt, 16 * mt, 128 + 32 * h, QK_LD, lane);  // K^T rows
    v16h bf = load_frag_b(QKt, 16 * nt, 32 * h, QK_LD, lane);        // Q^T rows
    v8f acc = {};
    acc = wmma_f16(a, bf, acc);
    const int q = 16 * nt + (lane & 15);
    _Float16* dst = Sh + ((size_t)h * 64 + q) * S_LD + 16 * mt + 8 * half;
    store_row8_f16(dst, acc, 1.0f);
  }
  __syncthreads();

  // ---- Stage 3: softmax over keys, one (head,row) per thread, fp32 math ----
  {
    const int h = tid >> 6, q = tid & 63;
    _Float16* row = Sh + ((size_t)h * 64 + q) * S_LD;
    float vals[64];
    float m = -1e30f;
#pragma unroll
    for (int j = 0; j < 64; ++j) { vals[j] = (float)row[j]; m = fmaxf(m, vals[j]); }
    float s = 0.f;
#pragma unroll
    for (int j = 0; j < 64; ++j) { vals[j] = __expf(vals[j] - m); s += vals[j]; }
    const float inv = 1.0f / s;
#pragma unroll
    for (int j = 0; j < 64; ++j) row[j] = (_Float16)(vals[j] * inv);
  }
  __syncthreads();

  // ---- Stage 4: O = V attn^T per head; M=channel(32), N=token; 4h x 2x4 ----
  // A = Vb rows (ch, m contiguous); B^T = attn[q][m] = Sh rows. D rows (M=ch)
  // contiguous in Ot[token][ch].
  for (int t = wave; t < 32; t += 8) {
    const int h  = t >> 3;
    const int mt = (t >> 2) & 1, nt = t & 3;     // mt: ch-tile (2), nt: token-tile
    const _Float16* sp = Sh + (size_t)h * 64 * S_LD;
    v8f acc = {};
#pragma unroll
    for (int ks = 0; ks < 2; ++ks) {
      v16h a  = load_frag_a(Vb, 32 * h + 16 * mt, 32 * ks, V_LD, lane);
      v16h bf = load_frag_b(sp, 16 * nt, 32 * ks, S_LD, lane);
      acc = wmma_f16(a, bf, acc);
    }
    const int tok = 16 * nt + (lane & 15);
    _Float16* dst = Ot + tok * OT_LD + 32 * h + 16 * mt + 8 * half;
    store_row8_f16(dst, acc, 1.0f);
  }
  __syncthreads();

  // ---- Stage 5: Y^T = O^T w_proj^T + bias; M=token, N=out-ch; 4x8 tiles ----
  // B^T = w_proj as stored ([o][c]). D rows (M=token) contiguous in
  // Yf[och][token] -> 2 x b128 store per lane.
  for (int t = wave; t < 32; t += 8) {
    const int nt = t >> 2, mt = t & 3;           // nt: och-tile 0..7, mt: token-tile
    v8f acc = {};
#pragma unroll
    for (int ks = 0; ks < 4; ++ks) {
      v16h a  = load_frag_a(Ot, 16 * mt, 32 * ks, OT_LD, lane);
      v16h bf = load_frag_b(wp16, 16 * nt, 32 * ks, 128, lane);
      acc = wmma_f16(a, bf, acc);
    }
    const int och = 16 * nt + (lane & 15);
    float* dst = Yf + och * Y_LD + 16 * mt + 8 * half;
    store_row8_f32(dst, acc, bsh[och]);
  }
  __syncthreads();

  // ---- Stage 6: coalesced float4 store of Yf -> out[b, c, h, w] ----
  float* ob = out + (size_t)b * 128 * 65536;
#pragma unroll
  for (int it = 0; it < 8; ++it) {
    const int q  = it * 256 + tid;
    const int c  = q >> 4;
    const int n0 = (q & 15) << 2;
    const int i = n0 >> 3, j0 = n0 & 7;
    float4 v;
    v.x = Yf[c * Y_LD + n0 + 0];
    v.y = Yf[c * Y_LD + n0 + 1];
    v.z = Yf[c * Y_LD + n0 + 2];
    v.w = Yf[c * Y_LD + n0 + 3];
    *(float4*)(ob + (size_t)c * 65536 + (size_t)(h0 + i) * 256 + (w0 + j0)) = v;
  }
}

extern "C" void kernel_launch(void* const* d_in, const int* in_sizes, int n_in,
                              void* d_out, int out_size, void* d_ws, size_t ws_size,
                              hipStream_t stream) {
  (void)in_sizes; (void)n_in; (void)out_size; (void)ws_size;
  const float* x      = (const float*)d_in[0];
  const float* w_qkv  = (const float*)d_in[1];
  const float* w_proj = (const float*)d_in[2];
  const float* b_proj = (const float*)d_in[3];
  float* out = (float*)d_out;

  _Float16* w16 = (_Float16*)d_ws;                 // [384*128] qkv, then [128*128] proj
  cvt_weights_kernel<<<192, 256, 0, stream>>>(w_qkv, w_proj, w16);

  win_attn_kernel<<<8192, 256, 0, stream>>>(x, w16, w16 + 384 * 128, b_proj, out);
}
